// CrossAlignedEnvEncoder_2602750181776
// MI455X (gfx1250) — compile-verified, matrix operand order checked by reading, not run on hardware
//
#include <hip/hip_runtime.h>
#include <hip/hip_bf16.h>

// ---------------------------------------------------------------------------
// CDNA5 (gfx1250, wave32) implementation of CrossAlignedEnvEncoder.
// All GEMMs and both attentions run on v_wmma_f32_16x16x32_bf16.
// GEMM is register-blocked 16x64 per wave (A fragment reused across 4 B tiles)
// to lift arithmetic intensity above the VMEM issue limit.
// ---------------------------------------------------------------------------

typedef __attribute__((ext_vector_type(16))) __bf16 v16bf;
typedef __attribute__((ext_vector_type(8)))  float  v8f;

union FragBF { v16bf v; uint4 q[2]; };

__device__ __forceinline__ unsigned short f2bf(float f) {
  unsigned int u = __float_as_uint(f);
  unsigned int r = u + 0x7FFFu + ((u >> 16) & 1u);   // round-to-nearest-even
  return (unsigned short)(r >> 16);
}

// ---------------------------------------------------------------------------
// Utility kernels
// ---------------------------------------------------------------------------
__global__ void k_zero_f32(float* p, int n) {
  int i = blockIdx.x * blockDim.x + threadIdx.x;
  if (i < n) p[i] = 0.0f;
}

__global__ void k_cast_bf16(const float* __restrict__ s, unsigned short* __restrict__ d, int n) {
  int i = blockIdx.x * blockDim.x + threadIdx.x;
  if (i < n) d[i] = f2bf(s[i]);
}

// d[c*R + r] = bf16(s[r*C + c])   (for weight_local used as x @ W)
__global__ void k_transpose_cast(const float* __restrict__ s, unsigned short* __restrict__ d,
                                 int R, int C) {
  int i = blockIdx.x * blockDim.x + threadIdx.x;
  if (i >= R * C) return;
  int r = i / C, c = i % C;
  d[(size_t)c * R + r] = f2bf(s[i]);
}

// ---------------------------------------------------------------------------
// GCN conv: deg[col] += 1 ; hi[col] += rsqrt(d[col])*rsqrt(d[row]) * x[row]
// ---------------------------------------------------------------------------
__global__ void k_degree(const int* __restrict__ ei, float* __restrict__ deg, int ne) {
  int e = blockIdx.x * blockDim.x + threadIdx.x;
  if (e < ne) atomicAdd(&deg[ei[ne + e]], 1.0f);
}

__global__ void k_gcn_scatter(const int* __restrict__ ei, const float* __restrict__ deg,
                              const float* __restrict__ x, float* __restrict__ hi, int ne) {
  int e = blockIdx.x;           // one edge per block
  int t = threadIdx.x;          // 256 features
  int r = ei[e];
  int c = ei[ne + e];
  float v = rsqrtf(deg[c]) * rsqrtf(deg[r]);
  if (!(v < 3.4e38f)) v = 0.0f;                     // kills inf/nan (nan_to_num)
  atomicAdd(&hi[(size_t)c * 256 + t], v * x[(size_t)r * 256 + t]);
}

// ---------------------------------------------------------------------------
// bf16 WMMA GEMM:  C[M,N] = A[M,K] @ B[N,K]^T + bias[N]  (f32 out)
// One 16(M) x 64(N) output block per wave: per K-step of 32 the A fragment is
// loaded once and reused across 4 B fragments -> 4 wmma per 10 b128 loads.
// N must be a multiple of 64, K a multiple of 32.
// ---------------------------------------------------------------------------
__global__ __launch_bounds__(128) void k_gemm_bf16(
    const unsigned short* __restrict__ A, int lda,
    const unsigned short* __restrict__ B, int ldb,
    const float* __restrict__ bias,
    float* __restrict__ C, int ldc,
    unsigned short* __restrict__ Cb, int ldcb,
    int M, int N, int K, int relu) {
  int wave = threadIdx.x >> 5;
  int lane = threadIdx.x & 31;
  int tile = blockIdx.x * 4 + wave;
  int ntn  = N >> 6;                                // N blocks of 64
  int tm = tile / ntn, tn = tile % ntn;
  if (tm >= (M >> 4)) return;                       // wave-uniform exit, EXEC stays all-1

  int r = lane & 15, half = lane >> 4, kb8 = half << 3;
  const unsigned short* arow = A + (size_t)(tm * 16 + r) * lda;
  const unsigned short* brow0 = B + (size_t)(tn * 64 + r) * ldb;

  v8f acc[4] = {};
  for (int k0 = 0; k0 < K; k0 += 32) {
    __builtin_prefetch(arow + k0 + 64, 0, 1);       // global_prefetch_b8
    FragBF af;
    af.q[0] = *(const uint4*)(arow + k0 + kb8);
    af.q[1] = *(const uint4*)(arow + k0 + 16 + kb8);
#pragma unroll
    for (int t = 0; t < 4; t++) {
      const unsigned short* brow = brow0 + (size_t)(t * 16) * ldb;
      FragBF bf;
      bf.q[0] = *(const uint4*)(brow + k0 + kb8);
      bf.q[1] = *(const uint4*)(brow + k0 + 16 + kb8);
      acc[t] = __builtin_amdgcn_wmma_f32_16x16x32_bf16(false, af.v, false, bf.v,
                                                       (short)0, acc[t], false, false);
    }
  }

  int row0 = tm * 16 + half * 8;
#pragma unroll
  for (int t = 0; t < 4; t++) {
    int col  = tn * 64 + t * 16 + r;
    float bv = bias ? bias[col] : 0.0f;
#pragma unroll
    for (int i = 0; i < 8; i++) {
      float v = acc[t][i] + bv;
      if (relu) v = v > 0.0f ? v : 0.0f;
      C[(size_t)(row0 + i) * ldc + col] = v;
      if (Cb) Cb[(size_t)(row0 + i) * ldcb + col] = f2bf(v);
    }
  }
}

// ---------------------------------------------------------------------------
// Pack qkv f32 [N,768] -> Qh,Kh bf16 [H][N][64] (scale folded into Q) and
// Vt bf16 [H][64][N] (V transposed so P@V B-fragments load contiguously).
// ---------------------------------------------------------------------------
__global__ void k_pack_qkv(const float* __restrict__ qkv,
                           unsigned short* __restrict__ Qh,
                           unsigned short* __restrict__ Kh,
                           unsigned short* __restrict__ Vt,
                           int N, float scale) {
  int gid = blockIdx.x * blockDim.x + threadIdx.x;
  if (gid >= N * 256) return;
  int n = gid >> 8, c = gid & 255;
  int h = c >> 6,  d = c & 63;
  const float* row = qkv + (size_t)n * 768;
  Qh[((size_t)h * N + n) * 64 + d] = f2bf(row[c] * scale);
  Kh[((size_t)h * N + n) * 64 + d] = f2bf(row[256 + c]);
  Vt[((size_t)h * 64 + d) * N + n] = f2bf(row[512 + c]);
}

// ---------------------------------------------------------------------------
// Flash attention (online softmax), one wave per (head, 16-query tile).
// Key blocks of 32: 4 wmma for S = Q@K^T, 4 wmma for O += P@V.
// P goes D-layout -> bf16 LDS tile -> A-layout (same-wave dscnt ordering).
// ---------------------------------------------------------------------------
__global__ __launch_bounds__(128) void k_flash_attn(
    const unsigned short* __restrict__ Qh,   // [H][Nq][64]
    const unsigned short* __restrict__ Kh,   // [H][Nk][64]
    const unsigned short* __restrict__ Vt,   // [H][64][Nk]
    float* __restrict__ O, int ldo,          // [Nq][H*64]
    int Nq, int Nk) {
  __shared__ __align__(16) unsigned short ldsP[4][16 * 32];

  int wave = threadIdx.x >> 5;
  int lane = threadIdx.x & 31;
  int tile = blockIdx.x * 4 + wave;
  int qtiles = Nq >> 4;
  int h  = tile / qtiles;
  int q0 = (tile % qtiles) << 4;
  int r = lane & 15, half = lane >> 4, kb8 = half << 3;

  // preload Q fragments (2 K-steps of 32)
  const unsigned short* qrow = Qh + ((size_t)h * Nq + q0 + r) * 64;
  FragBF qf[2];
#pragma unroll
  for (int ks = 0; ks < 2; ks++) {
    qf[ks].q[0] = *(const uint4*)(qrow + ks * 32 + kb8);
    qf[ks].q[1] = *(const uint4*)(qrow + ks * 32 + 16 + kb8);
  }

  v8f o[4] = {};
  float m[8], l[8];
#pragma unroll
  for (int i = 0; i < 8; i++) { m[i] = -3.0e38f; l[i] = 0.0f; }

  const unsigned short* kbp = Kh + (size_t)h * Nk * 64;
  const unsigned short* vbp = Vt + (size_t)h * 64 * Nk;
  unsigned short* lp = ldsP[wave];

  for (int kb = 0; kb < Nk; kb += 32) {
    // ---- S = Q @ K^T : two 16x16 column tiles ----
    v8f s0 = {}, s1 = {};
#pragma unroll
    for (int c = 0; c < 2; c++) {
      const unsigned short* krow = kbp + (size_t)(kb + c * 16 + r) * 64;
      FragBF k0f, k1f;
      k0f.q[0] = *(const uint4*)(krow + kb8);
      k0f.q[1] = *(const uint4*)(krow + 16 + kb8);
      k1f.q[0] = *(const uint4*)(krow + 32 + kb8);
      k1f.q[1] = *(const uint4*)(krow + 48 + kb8);
      v8f s = {};
      s = __builtin_amdgcn_wmma_f32_16x16x32_bf16(false, qf[0].v, false, k0f.v,
                                                  (short)0, s, false, false);
      s = __builtin_amdgcn_wmma_f32_16x16x32_bf16(false, qf[1].v, false, k1f.v,
                                                  (short)0, s, false, false);
      if (c == 0) s0 = s; else s1 = s;
    }

    // ---- online softmax (row reductions across the 16-lane groups) ----
    float p0[8], p1[8];
#pragma unroll
    for (int i = 0; i < 8; i++) {
      float mx = fmaxf(s0[i], s1[i]);
      mx = fmaxf(mx, __shfl_xor(mx, 1, 32));
      mx = fmaxf(mx, __shfl_xor(mx, 2, 32));
      mx = fmaxf(mx, __shfl_xor(mx, 4, 32));
      mx = fmaxf(mx, __shfl_xor(mx, 8, 32));
      float mn = fmaxf(m[i], mx);
      float alpha = __expf(m[i] - mn);
      m[i] = mn;
      float e0 = __expf(s0[i] - mn);
      float e1 = __expf(s1[i] - mn);
      float sum = e0 + e1;
      sum += __shfl_xor(sum, 1, 32);
      sum += __shfl_xor(sum, 2, 32);
      sum += __shfl_xor(sum, 4, 32);
      sum += __shfl_xor(sum, 8, 32);
      l[i] = l[i] * alpha + sum;
      o[0][i] *= alpha; o[1][i] *= alpha; o[2][i] *= alpha; o[3][i] *= alpha;
      p0[i] = e0; p1[i] = e1;
    }

    // ---- P (D-layout) -> bf16 LDS tile -> A-layout fragment ----
#pragma unroll
    for (int i = 0; i < 8; i++) {
      int rowp = half * 8 + i;
      lp[rowp * 32 + r]      = f2bf(p0[i]);
      lp[rowp * 32 + 16 + r] = f2bf(p1[i]);
    }
    FragBF pf;   // same-wave LDS RAW: compiler inserts s_wait_dscnt
    pf.q[0] = *(const uint4*)(lp + r * 32 + kb8);
    pf.q[1] = *(const uint4*)(lp + r * 32 + 16 + kb8);

    // ---- O += P @ V : four 16x16 column tiles over dh=64 ----
#pragma unroll
    for (int ct = 0; ct < 4; ct++) {
      const unsigned short* vrow = vbp + (size_t)(ct * 16 + r) * Nk + kb;
      FragBF vf;
      vf.q[0] = *(const uint4*)(vrow + kb8);
      vf.q[1] = *(const uint4*)(vrow + 16 + kb8);
      o[ct] = __builtin_amdgcn_wmma_f32_16x16x32_bf16(false, pf.v, false, vf.v,
                                                      (short)0, o[ct], false, false);
    }
  }

  // ---- epilogue: normalize and store ----
#pragma unroll
  for (int i = 0; i < 8; i++) {
    float inv = 1.0f / l[i];
    int row = q0 + half * 8 + i;
    float* orow = O + (size_t)row * ldo + h * 64;
    orow[r]      = o[0][i] * inv;
    orow[16 + r] = o[1][i] * inv;
    orow[32 + r] = o[2][i] * inv;
    orow[48 + r] = o[3][i] * inv;
  }
}

// ---------------------------------------------------------------------------
// y = LN(x + yadd) * g + b ; writes f32 (optional) and bf16 (optional, ld).
// One wave per row (E=256 -> 8 elements/lane).
// ---------------------------------------------------------------------------
__global__ __launch_bounds__(256) void k_add_ln(
    const float* __restrict__ X, const float* __restrict__ Y,
    const float* __restrict__ g, const float* __restrict__ b,
    float* __restrict__ outF, unsigned short* __restrict__ outB, int ldob,
    int N, int E) {
  int wave = threadIdx.x >> 5, lane = threadIdx.x & 31;
  int row = blockIdx.x * 8 + wave;
  if (row >= N) return;
  const float* x = X + (size_t)row * E;
  const float* y = Y + (size_t)row * E;
  float v[8], s = 0.0f, s2 = 0.0f;
#pragma unroll
  for (int i = 0; i < 8; i++) {
    float t = x[lane + 32 * i] + y[lane + 32 * i];
    v[i] = t; s += t; s2 += t * t;
  }
#pragma unroll
  for (int msk = 1; msk < 32; msk <<= 1) {
    s  += __shfl_xor(s,  msk, 32);
    s2 += __shfl_xor(s2, msk, 32);
  }
  float mean = s / E;
  float var  = s2 / E - mean * mean;
  float inv  = rsqrtf(var + 1e-5f);
#pragma unroll
  for (int i = 0; i < 8; i++) {
    int c = lane + 32 * i;
    float ov = (v[i] - mean) * inv * g[c] + b[c];
    if (outF) outF[(size_t)row * E + c] = ov;
    if (outB) outB[(size_t)row * ldob + c] = f2bf(ov);
  }
}

// ---------------------------------------------------------------------------
// Host orchestration
// ---------------------------------------------------------------------------
extern "C" void kernel_launch(void* const* d_in, const int* in_sizes, int n_in,
                              void* d_out, int out_size, void* d_ws, size_t ws_size,
                              hipStream_t stream) {
  const int N = 4096, E = 256, NE = 131072;
  const float scale = 0.125f;   // 1/sqrt(64)

  const float* x        = (const float*)d_in[0];
  const int*   ei       = (const int*)  d_in[1];
  const float* w_local  = (const float*)d_in[2];
  const float* sa_in_w  = (const float*)d_in[3];
  const float* sa_in_b  = (const float*)d_in[4];
  const float* sa_out_w = (const float*)d_in[5];
  const float* sa_out_b = (const float*)d_in[6];
  const float* ln1_g    = (const float*)d_in[7];
  const float* ln1_b    = (const float*)d_in[8];
  const float* ca_in_w  = (const float*)d_in[9];
  const float* ca_in_b  = (const float*)d_in[10];
  const float* ca_out_w = (const float*)d_in[11];
  const float* ca_out_b = (const float*)d_in[12];
  const float* ln2_g    = (const float*)d_in[13];
  const float* ln2_b    = (const float*)d_in[14];
  const float* gate_w   = (const float*)d_in[15];
  const float* gate_b   = (const float*)d_in[16];
  const float* fc_w     = (const float*)d_in[17];
  const float* fc_b     = (const float*)d_in[18];

  char* wp = (char*)d_ws;
  auto alloc = [&](size_t bytes) -> void* {
    void* p = wp; wp += (bytes + 255) & ~(size_t)255; return p;
  };
  typedef unsigned short us;
  float* deg     = (float*)alloc((size_t)N * 4);
  float* hi      = (float*)alloc((size_t)N * E * 4);
  us*    xbf     = (us*)   alloc((size_t)N * E * 2);
  us*    hibf    = (us*)   alloc((size_t)N * E * 2);
  us*    wlT     = (us*)   alloc((size_t)E * E * 2);
  us*    wsa_in  = (us*)   alloc((size_t)3 * E * E * 2);
  us*    wsa_out = (us*)   alloc((size_t)E * E * 2);
  us*    wca_in  = (us*)   alloc((size_t)3 * E * E * 2);
  us*    wca_out = (us*)   alloc((size_t)E * E * 2);
  us*    wgate   = (us*)   alloc((size_t)E * 2 * E * 2);
  us*    wfc     = (us*)   alloc((size_t)E * E * 2);
  float* localF  = (float*)alloc((size_t)N * E * 4);
  us*    localB  = (us*)   alloc((size_t)N * E * 2);
  float* qkvF    = (float*)alloc((size_t)N * 3 * E * 4);
  us*    Qh      = (us*)   alloc((size_t)N * E * 2);
  us*    KhB     = (us*)   alloc((size_t)N * E * 2);
  us*    VtB     = (us*)   alloc((size_t)N * E * 2);
  float* attF    = (float*)alloc((size_t)N * E * 4);
  us*    attB    = (us*)   alloc((size_t)N * E * 2);
  float* projF   = (float*)alloc((size_t)N * E * 4);
  float* geF     = (float*)alloc((size_t)N * E * 4);
  us*    concatB = (us*)   alloc((size_t)N * 2 * E * 2);
  float* gateF   = (float*)alloc((size_t)N * E * 4);
  us*    gateB   = (us*)   alloc((size_t)N * E * 2);

  auto gemm = [&](const us* A, int lda, const us* B, int ldb, const float* bias,
                  float* C, int ldc, us* Cb, int ldcb, int M, int Nn, int K, int relu) {
    int tiles = (M / 16) * (Nn / 64);               // 16x64 block per wave
    k_gemm_bf16<<<dim3((tiles + 3) / 4), dim3(128), 0, stream>>>(
        A, lda, B, ldb, bias, C, ldc, Cb, ldcb, M, Nn, K, relu);
  };
  auto cast = [&](const float* s, us* d, int n) {
    k_cast_bf16<<<dim3((n + 255) / 256), dim3(256), 0, stream>>>(s, d, n);
  };

  // ---- GCN conv ----
  k_zero_f32<<<dim3((N + 255) / 256), dim3(256), 0, stream>>>(deg, N);
  k_zero_f32<<<dim3((N * E + 255) / 256), dim3(256), 0, stream>>>(hi, N * E);
  k_degree<<<dim3((NE + 255) / 256), dim3(256), 0, stream>>>(ei, deg, NE);
  k_gcn_scatter<<<dim3(NE), dim3(256), 0, stream>>>(ei, deg, x, hi, NE);

  // ---- one-time casts ----
  cast(x, xbf, N * E);
  cast(hi, hibf, N * E);
  k_transpose_cast<<<dim3((E * E + 255) / 256), dim3(256), 0, stream>>>(w_local, wlT, E, E);
  cast(sa_in_w,  wsa_in,  3 * E * E);
  cast(sa_out_w, wsa_out, E * E);
  cast(ca_in_w,  wca_in,  3 * E * E);
  cast(ca_out_w, wca_out, E * E);
  cast(gate_w,   wgate,   E * 2 * E);
  cast(fc_w,     wfc,     E * E);

  // ---- local_embed = hi @ weight_local ----
  gemm(hibf, E, wlT, E, nullptr, localF, E, localB, E, N, E, E, 0);

  // ---- self-attention ----
  gemm(xbf, E, wsa_in, E, sa_in_b, qkvF, 3 * E, nullptr, 0, N, 3 * E, E, 0);
  k_pack_qkv<<<dim3((N * 256 + 255) / 256), dim3(256), 0, stream>>>(qkvF, Qh, KhB, VtB, N, scale);
  k_flash_attn<<<dim3((4 * (N / 16)) / 4), dim3(128), 0, stream>>>(Qh, KhB, VtB, attF, E, N, N);
  cast(attF, attB, N * E);
  gemm(attB, E, wsa_out, E, sa_out_b, projF, E, nullptr, 0, N, E, E, 0);

  // ---- ln1: global_embed (f32 + bf16 into concat[:, :256]) ----
  k_add_ln<<<dim3(N / 8), dim3(256), 0, stream>>>(x, projF, ln1_g, ln1_b,
                                                  geF, concatB, 2 * E, N, E);

  // ---- cross-attention: q from global_embed, k/v from local_embed ----
  gemm(concatB, 2 * E, wca_in, E, ca_in_b, qkvF, 3 * E, nullptr, 0, N, E, E, 0);
  gemm(localB, E, wca_in + (size_t)E * E, E, ca_in_b + E,
       qkvF + E, 3 * E, nullptr, 0, N, 2 * E, E, 0);
  k_pack_qkv<<<dim3((N * 256 + 255) / 256), dim3(256), 0, stream>>>(qkvF, Qh, KhB, VtB, N, scale);
  k_flash_attn<<<dim3((4 * (N / 16)) / 4), dim3(128), 0, stream>>>(Qh, KhB, VtB, attF, E, N, N);
  cast(attF, attB, N * E);
  gemm(attB, E, wca_out, E, ca_out_b, projF, E, nullptr, 0, N, E, E, 0);

  // ---- ln2: aligned -> bf16 into concat[:, 256:512] ----
  k_add_ln<<<dim3(N / 8), dim3(256), 0, stream>>>(geF, projF, ln2_g, ln2_b,
                                                  nullptr, concatB + E, 2 * E, N, E);

  // ---- gate: relu(concat @ gate_w^T + gate_b), K=512 ----
  gemm(concatB, 2 * E, wgate, 2 * E, gate_b, gateF, E, gateB, E, N, E, 2 * E, 1);

  // ---- fc: final output (f32) ----
  gemm(gateB, E, wfc, E, fc_b, (float*)d_out, E, nullptr, 0, N, E, E, 0);
}
